// GCNModel_31842887532566
// MI455X (gfx1250) — compile-verified, matrix operand order checked by reading, not run on hardware
//
#include <hip/hip_runtime.h>
#include <hip/hip_bf16.h>

// ---------------------------------------------------------------------------
// GCN on MI455X (gfx1250)
//   * bf16 WMMA GEMMs (v_wmma_f32_16x16x32_bf16), fp32 accumulate
//   * CSR (counting-sort by dst) built once per launch -> atomic-free gather
//   * gather fuses self-loop + bias + LayerNorm + residual + ReLU per wave
// ---------------------------------------------------------------------------

typedef __attribute__((ext_vector_type(16))) __bf16 v16bf;
typedef __attribute__((ext_vector_type(8)))  __bf16 v8bf;
typedef __attribute__((ext_vector_type(8)))  float  v8f;

#define HDIM 256

// ---------------- prep kernels ----------------------------------------------

__global__ __launch_bounds__(256) void k_init(float* deg, int* cnt, float* colsum, int N) {
    int i = blockIdx.x * 256 + threadIdx.x;
    if (i < N)   { deg[i] = 1.0f; cnt[i] = 0; }   // self-loop weight = 1
    if (i < 256) colsum[i] = 0.0f;
}

// weighted degree (float) + in-degree count (int), per edge
__global__ __launch_bounds__(256) void k_deg(const int* __restrict__ dst,
                                             const float* __restrict__ w,
                                             float* deg, int* cnt, int E) {
    int e = blockIdx.x * 256 + threadIdx.x;
    if (e < E) {
        int d = dst[e];
        atomicAdd(&deg[d], w[e]);
        atomicAdd(&cnt[d], 1);
    }
}

__global__ __launch_bounds__(256) void k_dinv(float* deg, int N) {
    int i = blockIdx.x * 256 + threadIdx.x;
    if (i < N) deg[i] = __frsqrt_rn(deg[i]);      // deg >= 1 always
}

// single-workgroup exclusive scan of cnt[0..N) -> row_start[0..N], cursor copy
__global__ __launch_bounds__(1024) void k_scan(const int* __restrict__ cnt,
                                               int* __restrict__ row_start,
                                               int* __restrict__ cursor, int N) {
    __shared__ int sbuf[1024];
    __shared__ int scarry;
    const int tid = threadIdx.x;
    if (tid == 0) scarry = 0;
    __syncthreads();
    for (int base = 0; base <= N; base += 1024) {
        int t = base + tid;
        int v = (t < N) ? cnt[t] : 0;
        sbuf[tid] = v;
        __syncthreads();
        int s = v;
        for (int o = 1; o < 1024; o <<= 1) {     // Hillis-Steele inclusive scan
            int add = (tid >= o) ? sbuf[tid - o] : 0;
            __syncthreads();
            s += add;
            sbuf[tid] = s;
            __syncthreads();
        }
        int carry = scarry;
        int excl  = carry + s - v;
        if (t <= N) {
            row_start[t] = excl;
            if (t < N) cursor[t] = excl;
        }
        __syncthreads();
        if (tid == 1023) scarry = carry + s;     // chunk total
        __syncthreads();
    }
}

// counting-sort fill: permuted (src, norm) arrays grouped by dst
__global__ __launch_bounds__(256) void k_build(const int* __restrict__ src,
                                               const int* __restrict__ dst,
                                               const float* __restrict__ w,
                                               const float* __restrict__ dinv,
                                               int* __restrict__ cursor,
                                               int* __restrict__ src_perm,
                                               float* __restrict__ norm_perm, int E) {
    int e = blockIdx.x * 256 + threadIdx.x;
    if (e >= E) return;
    int s = src[e], d = dst[e];
    float nm = dinv[s] * w[e] * dinv[d];
    int pos = atomicAdd(&cursor[d], 1);
    src_perm[pos]  = s;
    norm_perm[pos] = nm;
}

// Transpose + convert weights: WT[n*K + k] = (bf16) W[k*HDIM + n]
__global__ __launch_bounds__(256) void k_convW(const float* __restrict__ W,
                                               __bf16* __restrict__ WT, int K) {
    int idx = blockIdx.x * 256 + threadIdx.x;
    if (idx >= HDIM * K) return;
    int n = idx / K, k = idx % K;
    WT[idx] = (__bf16)W[k * HDIM + n];
}

// ---------------- WMMA GEMM: XW[N,256] = X[N,K] @ W[K,256] ------------------
__global__ __launch_bounds__(512) void k_gemm_bf16(const float* __restrict__ X,
                                                   const __bf16* __restrict__ WT,
                                                   float* __restrict__ XW, int K) {
    __shared__ alignas(16) __bf16 sA[16 * 256];
    const int row0 = blockIdx.x * 16;

    const int tot = 16 * K;
    for (int i = threadIdx.x; i < tot; i += 512)
        sA[i] = (__bf16)X[(size_t)row0 * K + i];
    __syncthreads();

    const int lane = threadIdx.x & 31;
    const int wid  = threadIdx.x >> 5;
    const int n0   = wid * 16;
    const int l15  = lane & 15;
    const int kh   = lane >> 4;

    v8f c = {};
    for (int kk = 0; kk < K; kk += 32) {
        const __bf16* pa = &sA[l15 * K + kk + kh * 8];
        v8bf alo = *(const v8bf*)pa;
        v8bf ahi = *(const v8bf*)(pa + 16);
        const __bf16* pb = &WT[(size_t)(n0 + l15) * K + kk + kh * 8];
        v8bf blo = *(const v8bf*)pb;
        v8bf bhi = *(const v8bf*)(pb + 16);

        v16bf a, b;
#pragma unroll
        for (int j = 0; j < 8; ++j) {
            a[j] = alo[j]; a[j + 8] = ahi[j];
            b[j] = blo[j]; b[j + 8] = bhi[j];
        }
        c = __builtin_amdgcn_wmma_f32_16x16x32_bf16(
                false, a, false, b, (short)0, c, false, false);
    }
#pragma unroll
    for (int r = 0; r < 8; ++r)
        XW[(size_t)(row0 + kh * 8 + r) * HDIM + n0 + l15] = c[r];
}

// ---------------- fused gather + epilogue -----------------------------------
// One wave per destination node; lane holds features [8*lane, 8*lane+8).
// MODE 0: x = acc + bias                      (first conv)
// MODE 1: x = relu(LN(acc + bias)*g + b + x)  (conv layers)
// MODE 2: MODE 1 + column-sum accumulation    (last conv layer)
template <int MODE>
__global__ __launch_bounds__(256) void k_gather(const float* __restrict__ xw,
                                                const int* __restrict__ row_start,
                                                const int* __restrict__ src_perm,
                                                const float* __restrict__ norm_perm,
                                                const float* __restrict__ dinv,
                                                const float* __restrict__ bias,
                                                const float* __restrict__ g,
                                                const float* __restrict__ bln,
                                                float* __restrict__ x,
                                                float* __restrict__ colsum, int N) {
    const int node = blockIdx.x * 8 + (threadIdx.x >> 5);
    const int lane = threadIdx.x & 31;
    if (node >= N) return;
    const int f0 = lane * 8;

    float acc[8];
    {   // self-loop contribution: xw[node] * dinv[node]^2
        float dn = dinv[node], sn = dn * dn;
        const float4* p = (const float4*)(xw + (size_t)node * HDIM + f0);
        float4 v0 = p[0], v1 = p[1];
        acc[0] = v0.x * sn; acc[1] = v0.y * sn; acc[2] = v0.z * sn; acc[3] = v0.w * sn;
        acc[4] = v1.x * sn; acc[5] = v1.y * sn; acc[6] = v1.z * sn; acc[7] = v1.w * sn;
    }

    const int beg = row_start[node], end = row_start[node + 1];
    for (int i = beg; i < end; ++i) {
        int   s  = src_perm[i];
        float nm = norm_perm[i];
        const float4* p = (const float4*)(xw + (size_t)s * HDIM + f0);
        float4 v0 = p[0], v1 = p[1];
        acc[0] += v0.x * nm; acc[1] += v0.y * nm; acc[2] += v0.z * nm; acc[3] += v0.w * nm;
        acc[4] += v1.x * nm; acc[5] += v1.y * nm; acc[6] += v1.z * nm; acc[7] += v1.w * nm;
    }

    float* xp = x + (size_t)node * HDIM + f0;
    if (MODE == 0) {
#pragma unroll
        for (int j = 0; j < 8; ++j) xp[j] = acc[j] + bias[f0 + j];
        return;
    }

    float h[8], s = 0.f, ss = 0.f;
#pragma unroll
    for (int j = 0; j < 8; ++j) {
        h[j] = acc[j] + bias[f0 + j];
        s  += h[j];
        ss += h[j] * h[j];
    }
#pragma unroll
    for (int o = 16; o >= 1; o >>= 1) {
        s  += __shfl_xor(s,  o, 32);
        ss += __shfl_xor(ss, o, 32);
    }
    float mu  = s * (1.0f / HDIM);
    float var = ss * (1.0f / HDIM) - mu * mu;
    float inv = __frsqrt_rn(var + 1e-5f);
#pragma unroll
    for (int j = 0; j < 8; ++j) {
        int f = f0 + j;
        float y = (h[j] - mu) * inv * g[f] + bln[f];
        float r = fmaxf(y + xp[j], 0.0f);
        xp[j] = r;
        if (MODE == 2) atomicAdd(&colsum[f], r);
    }
}

// out[j] = (colsum/N) . fc_W[:,j] + fc_b[j]
__global__ __launch_bounds__(256) void k_final(const float* __restrict__ colsum,
                                               const float* __restrict__ fcW,
                                               const float* __restrict__ fcb,
                                               float* __restrict__ out, float invN) {
    __shared__ float s0[256], s1[256];
    int t = threadIdx.x;
    float c = colsum[t] * invN;
    s0[t] = c * fcW[t * 2 + 0];
    s1[t] = c * fcW[t * 2 + 1];
    __syncthreads();
    for (int o = 128; o > 0; o >>= 1) {
        if (t < o) { s0[t] += s0[t + o]; s1[t] += s1[t + o]; }
        __syncthreads();
    }
    if (t == 0) { out[0] = s0[0] + fcb[0]; out[1] = s1[0] + fcb[1]; }
}

// ---------------------------------------------------------------------------

extern "C" void kernel_launch(void* const* d_in, const int* in_sizes, int n_in,
                              void* d_out, int out_size, void* d_ws, size_t ws_size,
                              hipStream_t stream) {
    const int N = 50000, E = 1600000, F = 128, H = HDIM, L = 3;

    const float* node  = (const float*)d_in[0];
    const int*   edges = (const int*)d_in[1];   // harness ABI: integer -> int*
    const int*   src   = edges;
    const int*   dst   = edges + E;
    const float* eattr = (const float*)d_in[2];
    const float* W1    = (const float*)d_in[3];
    const float* b1    = (const float*)d_in[4];
    const float* Wc    = (const float*)d_in[5];
    const float* bc    = (const float*)d_in[6];
    const float* lng   = (const float*)d_in[7];
    const float* lnb   = (const float*)d_in[8];
    const float* fcW   = (const float*)d_in[9];
    const float* fcb   = (const float*)d_in[10];
    float*       out   = (float*)d_out;

    char* base = (char*)d_ws;
    size_t off = 0;
    auto carve = [&](size_t bytes) {
        char* p = base + off;
        off += (bytes + 255) & ~(size_t)255;
        return p;
    };
    float*  dinv      = (float*)carve((size_t)N * 4);         // deg -> rsqrt in place
    int*    cnt       = (int*)carve((size_t)N * 4);
    int*    row_start = (int*)carve((size_t)(N + 1) * 4);
    int*    cursor    = (int*)carve((size_t)N * 4);
    int*    src_perm  = (int*)carve((size_t)E * 4);
    float*  norm_perm = (float*)carve((size_t)E * 4);
    __bf16* W1T       = (__bf16*)carve((size_t)H * F * 2);
    __bf16* WcT       = (__bf16*)carve((size_t)L * H * H * 2);
    float*  xbuf      = (float*)carve((size_t)N * H * 4);
    float*  xw        = (float*)carve((size_t)N * H * 4);
    float*  colsum    = (float*)carve(256 * 4);

    // --- graph-invariant prep: degrees, CSR build (once per launch) ---
    k_init <<<(N + 255) / 256, 256, 0, stream>>>(dinv, cnt, colsum, N);
    k_deg  <<<(E + 255) / 256, 256, 0, stream>>>(dst, eattr, dinv, cnt, E);
    k_dinv <<<(N + 255) / 256, 256, 0, stream>>>(dinv, N);
    k_scan <<<1, 1024, 0, stream>>>(cnt, row_start, cursor, N);
    k_build<<<(E + 255) / 256, 256, 0, stream>>>(src, dst, eattr, dinv, cursor,
                                                 src_perm, norm_perm, E);

    k_convW<<<(H * F) / 256, 256, 0, stream>>>(W1, W1T, F);
    for (int l = 0; l < L; ++l)
        k_convW<<<(H * H) / 256, 256, 0, stream>>>(Wc + (size_t)l * H * H,
                                                   WcT + (size_t)l * H * H, H);

    // --- conv1: node [N,F] -> x [N,H] ---
    k_gemm_bf16<<<N / 16, 512, 0, stream>>>(node, W1T, xw, F);
    k_gather<0><<<N / 8, 256, 0, stream>>>(xw, row_start, src_perm, norm_perm,
                                           dinv, b1, nullptr, nullptr, xbuf,
                                           nullptr, N);

    // --- 3 message-passing layers ---
    for (int i = 0; i < L; ++i) {
        k_gemm_bf16<<<N / 16, 512, 0, stream>>>(xbuf, WcT + (size_t)i * H * H, xw, H);
        if (i == L - 1)
            k_gather<2><<<N / 8, 256, 0, stream>>>(xw, row_start, src_perm, norm_perm,
                                                   dinv, bc + (size_t)i * H,
                                                   lng + (size_t)i * H, lnb + (size_t)i * H,
                                                   xbuf, colsum, N);
        else
            k_gather<1><<<N / 8, 256, 0, stream>>>(xw, row_start, src_perm, norm_perm,
                                                   dinv, bc + (size_t)i * H,
                                                   lng + (size_t)i * H, lnb + (size_t)i * H,
                                                   xbuf, colsum, N);
    }

    // --- FC + global mean pool ---
    k_final<<<1, 256, 0, stream>>>(colsum, fcW, fcb, out, 1.0f / N);
}
